// LSTMWithGaussianAttention_12103217840786
// MI455X (gfx1250) — matrix operand order repaired and checked by reading, not compiled, not run on hardware
//
#include <hip/hip_runtime.h>

// ---------------------------------------------------------------------------
// Graves attention-LSTM, persistent 16-workgroup kernel for MI455X (gfx1250).
//  * Fused weight Wcat[2048][640] f16 resident in LDS (160KB slice / WGP).
//  * Per-step GEMM via v_wmma_f32_16x16x32_f16, software-pipelined LDS loads.
//  * X stored A-swizzled in LDS so A- and B-fragments are 2x ds_load_b128 each.
//  * Activations via hardware v_tanh_f32 (CDNA5 TRANS op).
//  * Cross-WGP h exchange via L2 staging + cluster barrier (s_barrier -3)
//    backed by a monotonic global atomic barrier (portable fallback).
// ---------------------------------------------------------------------------

typedef __attribute__((ext_vector_type(16))) _Float16 v16h;
typedef __attribute__((ext_vector_type(8)))  float    v8f;

#define BSZ   64
#define TLEN  512
#define HDIM  512
#define KGAUSS 10
#define NCHAR 80
#define ULEN  64
#define NBLK  16
#define NTHR  1024
#define KTOT  640      // 512 h | 3 x | 80 w | 45 zero-pad
#define NPW   128      // gate rows per WG = 32 units x {i,f,g,o}

// ---- LDS layout (bytes) ----------------------------------------------------
#define OFF_W   0
#define SZ_W    (NPW*KTOT*2)            // 163840  weight slice (f16)
#define OFF_X   (OFF_W + SZ_W)          // 163840
#define SZ_X    (BSZ*KTOT*2)            // 81920   X = [h|x|w] (f16, A-swizzled)
#define OFF_S   (OFF_X + SZ_X)          // 245760  scratch union:
#define SZ_S    33792                   //   Gl 64x128 f32  OR  pbuf/abuf/bbuf/wbuf
#define OFF_C   (OFF_S + SZ_S)          // 279552  cell state 64x32 f32
#define SZ_C    (BSZ*32*4)              // 8192
#define OFF_K   (OFF_C + SZ_C)          // 287744  kappa 64x10 f32
#define SZ_K    (BSZ*KGAUSS*4)          // 2560
#define OFF_WW  (OFF_K + SZ_K)          // 290304  W_win padded 32x512 f16
#define SZ_WW   (32*HDIM*2)             // 32768
#define OFF_B   (OFF_WW + SZ_WW)        // 323072  bias slice 128 f32
#define SZ_B    (NPW*4)                 // 512
#define SMEM_TOTAL (OFF_B + SZ_B)       // 323584  (< 320KB WGP LDS)

// ---- workspace layout (bytes) ---------------------------------------------
#define WS_CTR   0
#define WS_XG    1024                         // h staging: 64x512 f16 = 64KB
#define WS_WPACK (WS_XG + BSZ*HDIM*2)         // 16x128x640 f16 = 2621440
#define WS_WWIN  (WS_WPACK + NBLK*NPW*KTOT*2) // 32x512 f16
#define WS_BPACK (WS_WWIN + 32*HDIM*2)        // 2048 f32

union ABfrag { unsigned u[8]; uint4 q[2]; v16h h; };

// 16 contiguous halves -> one fragment (two ds_load_b128).
__device__ __forceinline__ v16h load_frag16(const _Float16* p) {
  ABfrag u;
  u.q[0] = *(const uint4*)(const void*)(p);
  u.q[1] = *(const uint4*)(const void*)(p + 8);
  return u.h;
}

// A-matrix swizzle (ISA 7.12.2): lane wants K = hi*8 + {0..7, 16..23} of each
// 32-half k-block. Store X with 8-half groups permuted [0,1,2,3]->[0,2,1,3],
// so lane (l,hi) reads 16 contiguous halves at block + hi*16.
__device__ __forceinline__ int swz(int k) {
  int g = (k >> 3) & 3;
  return (k & ~31) | ((((g & 1) << 1) | (g >> 1)) << 3) | (k & 7);
}

// CDNA5 hardware tanh (TRANS op). Exact sigmoid via tanh identity.
__device__ __forceinline__ float fast_tanh(float x) {
  float r;
  asm("v_tanh_f32 %0, %1" : "=v"(r) : "v"(x));
  return r;
}
__device__ __forceinline__ float fast_sigmoid(float x) {
  return fmaf(0.5f, fast_tanh(0.5f * x), 0.5f);
}

__device__ __forceinline__ void grid_barrier(unsigned* ctr, unsigned target) {
  __syncthreads();
  if (threadIdx.x == 0) {
    __threadfence();                       // release h stores
    atomicAdd(ctr, 1u);
    while (*(volatile unsigned*)ctr < target) __builtin_amdgcn_s_sleep(2);
    __threadfence();                       // acquire
  }
  // CDNA5 cluster user-barrier (s_barrier_signal/-wait -3). NOP when the
  // dispatch is not clustered (ClusterID==0), hardware sync when it is.
  __builtin_amdgcn_s_cluster_barrier();
  __syncthreads();
}

// ---------------------------------------------------------------------------
// Setup: pack W_hh|W_ih into per-WG f16 slices, pad W_win, fuse biases.
// ---------------------------------------------------------------------------
__global__ void lstm_attn_setup(const float* __restrict__ W_ih,
                                const float* __restrict__ W_hh,
                                const float* __restrict__ b_ih,
                                const float* __restrict__ b_hh,
                                const float* __restrict__ W_win,
                                _Float16* __restrict__ Wpack,
                                _Float16* __restrict__ WwinP,
                                float* __restrict__ bpack) {
  int idx = blockIdx.x * blockDim.x + threadIdx.x;
  const int NW = NBLK * NPW * KTOT;
  if (idx < NW) {
    int k  = idx % KTOT;
    int rr = idx / KTOT;
    int r  = rr % NPW, wg = rr / NPW;
    int gate = r >> 5, unit = wg * 32 + (r & 31);
    int gr = gate * HDIM + unit;              // global gate row in [0,2048)
    float v = 0.f;
    if (k < HDIM)          v = W_hh[(size_t)gr * HDIM + k];
    else if (k < HDIM + 3) v = W_ih[(size_t)gr * 83 + (k - HDIM)];
    else if (k < HDIM + 3 + NCHAR) v = W_ih[(size_t)gr * 83 + 3 + (k - HDIM - 3)];
    Wpack[rr * KTOT + swz(k)] = (_Float16)v;
  } else if (idx < NW + 32 * HDIM) {
    int j = idx - NW;
    int k = j % HDIM, r = j / HDIM;
    WwinP[r * HDIM + swz(k)] = (_Float16)((r < 3 * KGAUSS) ? W_win[r * HDIM + k] : 0.f);
  } else if (idx < NW + 32 * HDIM + NBLK * NPW) {
    int j = idx - NW - 32 * HDIM;
    int r = j % NPW, wg = j / NPW;
    int gate = r >> 5, unit = wg * 32 + (r & 31);
    int gr = gate * HDIM + unit;
    bpack[j] = b_ih[gr] + b_hh[gr];
  }
}

__global__ void lstm_attn_init(unsigned* ctr) { *ctr = 0u; }

// ---------------------------------------------------------------------------
// Main persistent kernel: 16 blocks x 1024 threads (32 waves, wave32).
// ---------------------------------------------------------------------------
__global__ __launch_bounds__(NTHR, 1)
void lstm_attn_main(const float* __restrict__ strokes,
                    const int*   __restrict__ sent,
                    const float* __restrict__ smask,
                    const float* __restrict__ b_win,
                    const _Float16* __restrict__ Wpack,
                    const _Float16* __restrict__ WwinP,
                    const float* __restrict__ bpack,
                    _Float16* __restrict__ Xg,
                    unsigned* __restrict__ ctr,
                    float* __restrict__ hs_out,
                    float* __restrict__ ws_out) {
  extern __shared__ char smem[];
  _Float16* Wl   = (_Float16*)(smem + OFF_W);
  _Float16* Xl   = (_Float16*)(smem + OFF_X);         // A-swizzled
  float*    Gl   = (float*)(smem + OFF_S);            // 64x128 gates
  float*    pbuf = (float*)(smem + OFF_S);            // 64x32 (overlays Gl)
  float*    abuf = (float*)(smem + OFF_S + 8192);     // 64x10
  float*    bbuf = (float*)(smem + OFF_S + 10752);    // 64x10
  float*    wbuf = (float*)(smem + OFF_S + 13312);    // 64x80
  float*    Cl   = (float*)(smem + OFF_C);            // 64x32 cell state
  float*    kap  = (float*)(smem + OFF_K);            // 64x10 kappa
  _Float16* Wwl  = (_Float16*)(smem + OFF_WW);        // 32x512 W_win
  float*    barr = (float*)(smem + OFF_B);            // 128 bias

  const int tid  = threadIdx.x;
  const int wg   = blockIdx.x;         // 0..15, owns hidden units [wg*32, wg*32+32)
  const int lane = tid & 31;
  const int wave = tid >> 5;
  const int l    = lane & 15;
  const int hi   = lane >> 4;

  // ---- one-time init: stage weight slice + W_win into LDS, zero state ----
  {
    const uint4* s = (const uint4*)(const void*)(Wpack + (size_t)wg * NPW * KTOT);
    uint4* d = (uint4*)(void*)Wl;
    for (int i = tid; i < NPW * KTOT / 8; i += NTHR) d[i] = s[i];
    const uint4* s2 = (const uint4*)(const void*)WwinP;
    uint4* d2 = (uint4*)(void*)Wwl;
    for (int i = tid; i < 32 * HDIM / 8; i += NTHR) d2[i] = s2[i];
    if (tid < NPW) barr[tid] = bpack[wg * NPW + tid];
    for (int i = tid; i < BSZ * KTOT; i += NTHR) Xl[i] = (_Float16)0.f;
    for (int i = tid; i < BSZ * 32; i += NTHR) Cl[i] = 0.f;
    for (int i = tid; i < BSZ * KGAUSS; i += NTHR) kap[i] = 0.f;
  }
  __syncthreads();
  for (int i = tid; i < BSZ * 3; i += NTHR) {       // x_0
    int b = i / 3, f = i % 3;
    Xl[b * KTOT + swz(HDIM + f)] = (_Float16)strokes[(size_t)(b * TLEN) * 3 + f];
  }
  __syncthreads();

  const int m0 = (wave >> 3) << 4;   // batch tile
  const int n0 = (wave & 7) << 4;    // gate-row tile within the 128-slice
  const float bn_bias = barr[n0 + l];   // loop-invariant bias for D column
  unsigned barn = 0;

  for (int t = 0; t < TLEN; ++t) {
    // ---- GEMM: G[64x128] = X[64x640] @ Wslice^T, pipelined WMMA ----
    const _Float16* xrow = Xl + (m0 + l) * KTOT + hi * 16;  // A (swizzled)
    const _Float16* wrow = Wl + (n0 + l) * KTOT + hi * 16;  // B
    v8f acc = {};
    v16h a = load_frag16(xrow);
    v16h b = load_frag16(wrow);
#pragma unroll 2
    for (int kt = 0; kt < KTOT / 32 - 1; ++kt) {
      v16h an = load_frag16(xrow + (kt + 1) * 32);
      v16h bn = load_frag16(wrow + (kt + 1) * 32);
      acc = __builtin_amdgcn_wmma_f32_16x16x32_f16(
          false, a, false, b, (short)0, acc, false, false);
      a = an; b = bn;
    }
    acc = __builtin_amdgcn_wmma_f32_16x16x32_f16(
        false, a, false, b, (short)0, acc, false, false);
#pragma unroll
    for (int v = 0; v < 8; ++v)                       // D layout: m=v+8*hi, n=lane
      Gl[(m0 + v + hi * 8) * NPW + (n0 + l)] = acc[v] + bn_bias;
    __syncthreads();

    // ---- LSTM pointwise for our 32 hidden units (v_tanh_f32) ----
    for (int q = tid; q < BSZ * 32; q += NTHR) {
      int b = q >> 5, j = q & 31;
      float gi = Gl[b * NPW + j];
      float gf = Gl[b * NPW + 32 + j];
      float gg = Gl[b * NPW + 64 + j];
      float go = Gl[b * NPW + 96 + j];
      float ii = fast_sigmoid(gi);
      float ff = fast_sigmoid(gf);
      float gt = fast_tanh(gg);
      float oo = fast_sigmoid(go);
      float c  = ff * Cl[q] + ii * gt;
      Cl[q] = c;
      float h = oo * fast_tanh(c);
      int col = wg * 32 + j;
      Xg[b * HDIM + col] = (_Float16)h;               // broadcast via L2
      hs_out[((size_t)b * TLEN + t) * HDIM + col] = h;
    }
    // prefetch next stroke row while waiting at the barrier
    if (t + 1 < TLEN && tid == 0)
      __builtin_prefetch(&strokes[(size_t)(t + 1) * 3], 0, 1);
    grid_barrier(ctr, (++barn) * NBLK);               // h of step t visible

    // ---- pull full h into local X (apply A-swizzle: groups [0,1,2,3]->[0,2,1,3]) ----
    {
      const uint4* s = (const uint4*)(const void*)Xg;
      for (int i = tid; i < BSZ * HDIM / 8; i += NTHR) {
        int b = i >> 6, c8 = i & 63;
        int g = c8 & 3;
        int c8p = (c8 & ~3) | ((g & 1) << 1) | (g >> 1);
        *(uint4*)(void*)(Xl + b * KTOT + c8p * 8) = s[i];
      }
    }
    grid_barrier(ctr, (++barn) * NBLK);               // Xg fully consumed

    // ---- window projection p = h @ W_win^T via WMMA (waves 0..7) ----
    if (wave < 8) {
      int pm0 = (wave >> 1) << 4;
      int pn0 = (wave & 1) << 4;
      const _Float16* pxrow = Xl + (pm0 + l) * KTOT + hi * 16;
      const _Float16* pwrow = Wwl + (pn0 + l) * HDIM + hi * 16;
      v8f pacc = {};
      v16h pa = load_frag16(pxrow);
      v16h pb = load_frag16(pwrow);
#pragma unroll 2
      for (int kt = 0; kt < HDIM / 32 - 1; ++kt) {
        v16h an = load_frag16(pxrow + (kt + 1) * 32);
        v16h bn = load_frag16(pwrow + (kt + 1) * 32);
        pacc = __builtin_amdgcn_wmma_f32_16x16x32_f16(
            false, pa, false, pb, (short)0, pacc, false, false);
        pa = an; pb = bn;
      }
      pacc = __builtin_amdgcn_wmma_f32_16x16x32_f16(
          false, pa, false, pb, (short)0, pacc, false, false);
#pragma unroll
      for (int v = 0; v < 8; ++v)
        pbuf[(pm0 + v + hi * 8) * 32 + (pn0 + l)] = pacc[v];
    }
    for (int i = tid; i < BSZ * NCHAR; i += NTHR) wbuf[i] = 0.f;
    __syncthreads();

    // ---- alpha/beta/kappa (all WGs redundantly; kappa replicated) ----
    if (tid < BSZ * KGAUSS) {
      int b = tid / KGAUSS, k = tid % KGAUSS;
      abuf[tid] = __expf(pbuf[b * 32 + k] + b_win[k]);
      bbuf[tid] = __expf(pbuf[b * 32 + KGAUSS + k] + b_win[KGAUSS + k]);
      kap[tid] += __expf(pbuf[b * 32 + 2 * KGAUSS + k] + b_win[2 * KGAUSS + k]);
    }
    __syncthreads();

    // ---- phi + one-hot scatter: w[b, sent[b,u]] += phi[b,u]*mask[b,u] ----
    for (int q = tid; q < BSZ * ULEN; q += NTHR) {
      int b = q >> 6, uu = q & 63;
      float s = 0.f;
#pragma unroll
      for (int k = 0; k < KGAUSS; ++k) {
        float d = kap[b * KGAUSS + k] - (float)uu;
        s += abuf[b * KGAUSS + k] * __expf(-bbuf[b * KGAUSS + k] * d * d);
      }
      float m = smask[b * ULEN + uu];
      int ch = sent[b * ULEN + uu];
      atomicAdd(&wbuf[b * NCHAR + ch], s * m);
    }
    __syncthreads();

    // ---- commit w into X (and output), stage x_{t+1} ----
    for (int q = tid; q < BSZ * NCHAR; q += NTHR) {
      int b = q / NCHAR, n = q % NCHAR;
      float v = wbuf[q];
      Xl[b * KTOT + swz(HDIM + 3 + n)] = (_Float16)v;
      if (wg == 0) ws_out[((size_t)b * TLEN + t) * NCHAR + n] = v;
    }
    if (t + 1 < TLEN) {
      for (int i = tid; i < BSZ * 3; i += NTHR) {
        int b = i / 3, f = i % 3;
        Xl[b * KTOT + swz(HDIM + f)] =
            (_Float16)strokes[((size_t)b * TLEN + (t + 1)) * 3 + f];
      }
    }
    __syncthreads();
  }
}

// ---------------------------------------------------------------------------
extern "C" void kernel_launch(void* const* d_in, const int* in_sizes, int n_in,
                              void* d_out, int out_size, void* d_ws, size_t ws_size,
                              hipStream_t stream) {
  const float* strokes = (const float*)d_in[0];
  const int*   sent    = (const int*)d_in[1];
  const float* smask   = (const float*)d_in[2];
  const float* W_ih    = (const float*)d_in[3];
  const float* W_hh    = (const float*)d_in[4];
  const float* b_ih    = (const float*)d_in[5];
  const float* b_hh    = (const float*)d_in[6];
  const float* W_win   = (const float*)d_in[7];
  const float* b_win   = (const float*)d_in[8];

  char* ws = (char*)d_ws;
  unsigned* ctr    = (unsigned*)(ws + WS_CTR);
  _Float16* Xg     = (_Float16*)(ws + WS_XG);
  _Float16* Wpack  = (_Float16*)(ws + WS_WPACK);
  _Float16* WwinP  = (_Float16*)(ws + WS_WWIN);
  float*    bpack  = (float*)(ws + WS_BPACK);

  lstm_attn_init<<<1, 1, 0, stream>>>(ctr);

  int total = NBLK * NPW * KTOT + 32 * HDIM + NBLK * NPW;
  lstm_attn_setup<<<(total + 255) / 256, 256, 0, stream>>>(
      W_ih, W_hh, b_ih, b_hh, W_win, Wpack, WwinP, bpack);

  float* hs_out = (float*)d_out;
  float* ws_out = hs_out + (size_t)BSZ * TLEN * HDIM;
  lstm_attn_main<<<NBLK, NTHR, SMEM_TOTAL, stream>>>(
      strokes, sent, smask, b_win, Wpack, WwinP, bpack, Xg, ctr, hs_out, ws_out);
}
